// MultiScaleDecoderFixed_31086973288691
// MI455X (gfx1250) — matrix operand order repaired and checked by reading, not compiled, not run on hardware
//
#include <hip/hip_runtime.h>
#include <hip/hip_bf16.h>
#include <math.h>

typedef __attribute__((ext_vector_type(16))) _Float16 v16h;
typedef __attribute__((ext_vector_type(8)))  float    v8f;
typedef __attribute__((ext_vector_type(4)))  float    v4f;
typedef __attribute__((ext_vector_type(4)))  int      v4i;

#define NC 1024      // coarse tokens per batch
#define NF 4096      // fine tokens per batch
#define NTOK 5120    // Nc + Nf
#define E 256        // embed dim
#define IMG 256      // Hc*8 = Hf*4 = 256
#define BATCH 32

// ---------------------------------------------------------------------------
// Kernel 0: pre-pack W_coarse / W_fine into per-lane f16 WMMA B-fragments.
// B fragment (32x16 f16): lane = {half = K-block, n = N}, element e -> K =
// half*16 + e.  Coarse: 8 ksteps x 4 ntiles, Fine: 8 ksteps x 1 ntile.
// coarse layout: ((ks*4+nt)*32 + lane)*16 + e
// fine   layout: ( ks      *32 + lane)*16 + e
// ---------------------------------------------------------------------------
__global__ void __launch_bounds__(256)
pack_bfrag_kernel(const float* __restrict__ Wc,
                  const float* __restrict__ Wf,
                  _Float16* __restrict__ fragC,
                  _Float16* __restrict__ fragF) {
    const int idx = blockIdx.x * 256 + threadIdx.x;   // 0 .. 20479
    if (idx < 8 * 4 * 32 * 16) {                      // coarse: 16384
        const int e    = idx & 15;
        const int lane = (idx >> 4) & 31;
        const int nt   = (idx >> 9) & 3;
        const int ks   = idx >> 11;
        const int k    = ks * 32 + (lane >> 4) * 16 + e;
        const int n    = nt * 16 + (lane & 15);
        fragC[idx] = (_Float16)Wc[(size_t)k * 64 + n];
    } else {                                          // fine: 4096
        const int fi   = idx - 8 * 4 * 32 * 16;
        const int e    = fi & 15;
        const int lane = (fi >> 4) & 31;
        const int ks   = fi >> 9;
        const int k    = ks * 32 + (lane >> 4) * 16 + e;
        const int n    = lane & 15;
        fragF[fi] = (_Float16)Wf[(size_t)k * 16 + n];
    }
}

// Load one A fragment (16x32 f16 tile row for this lane) from an fp32 token
// row: one contiguous 16-float run, fetched as 4 non-temporal b128 loads.
__device__ __forceinline__ v16h load_a_frag(const float* __restrict__ Arow,
                                            int kk, int half) {
    const v4f c0 = __builtin_nontemporal_load((const v4f*)(Arow + kk + half * 16));
    const v4f c1 = __builtin_nontemporal_load((const v4f*)(Arow + kk + half * 16 + 4));
    const v4f c2 = __builtin_nontemporal_load((const v4f*)(Arow + kk + half * 16 + 8));
    const v4f c3 = __builtin_nontemporal_load((const v4f*)(Arow + kk + half * 16 + 12));
    v16h a;
#pragma unroll
    for (int i = 0; i < 4; ++i) {
        a[i]      = (_Float16)c0[i];
        a[4 + i]  = (_Float16)c1[i];
        a[8 + i]  = (_Float16)c2[i];
        a[12 + i] = (_Float16)c3[i];
    }
    return a;
}

// ---------------------------------------------------------------------------
// Kernel 1: coarse GEMM  (B*Nc = 32768 rows) x 256 @ W_coarse(256x64) + b
// One wave computes a 16(M) x 64(N) tile via 4 WMMA accumulators, K in 8
// steps of 32.  Result scattered directly into pre-conv coarse image layout.
// ---------------------------------------------------------------------------
__global__ void __launch_bounds__(256)
coarse_gemm_kernel(const float* __restrict__ tokens,
                   const _Float16* __restrict__ fragC,
                   const float* __restrict__ bc,
                   float* __restrict__ coarse_img) {
    const int wave = threadIdx.x >> 5;
    const int lane = threadIdx.x & 31;
    const int half = lane >> 4;
    const int mloc = lane & 15;
    const int tile_m = blockIdx.x * 8 + wave;   // 2048 tiles total

    const int trow = tile_m * 16 + mloc;        // [0, 32768)
    const int b    = trow >> 10;
    const int pos  = trow & 1023;
    const float* Arow = tokens + ((size_t)b * NTOK + pos) * E;

    v8f acc[4] = {};

    for (int ks = 0; ks < 8; ++ks) {
        const v16h a = load_a_frag(Arow, ks * 32, half);
        const _Float16* fb = fragC + ((size_t)(ks * 4) * 32 + lane) * 16;
#pragma unroll
        for (int nt = 0; nt < 4; ++nt) {
            const v16h bf = *(const v16h*)(fb + (size_t)nt * 32 * 16);
            acc[nt] = __builtin_amdgcn_wmma_f32_16x16x32_f16(
                false, a, false, bf, (short)0, acc[nt], false, false);
        }
    }

    // scatter into pre-conv coarse image: pixel-shuffle 8x8
#pragma unroll
    for (int nt = 0; nt < 4; ++nt) {
        const int n  = nt * 16 + mloc;   // patch pixel id, dy*8+dx
        const int dy = n >> 3;
        const int dx = n & 7;
        const float bias = bc[n];
#pragma unroll
        for (int r = 0; r < 8; ++r) {
            const int M  = r + half * 8;
            const int t  = tile_m * 16 + M;
            const int bb = t >> 10;
            const int pp = t & 1023;
            const int hc = pp >> 5;
            const int wc = pp & 31;
            coarse_img[((size_t)bb * IMG + (hc * 8 + dy)) * IMG + (wc * 8 + dx)]
                = acc[nt][r] + bias;
        }
    }
}

// ---------------------------------------------------------------------------
// Kernel 2: fine GEMM  (B*Nf = 131072 rows) x 256 @ W_fine(256x16) + b
// One wave per 16x16 tile; result -> fine_all[b][j][16] (bias included).
// ---------------------------------------------------------------------------
__global__ void __launch_bounds__(256)
fine_gemm_kernel(const float* __restrict__ tokens,
                 const _Float16* __restrict__ fragF,
                 const float* __restrict__ bf_,
                 float* __restrict__ fine_all) {
    const int wave = threadIdx.x >> 5;
    const int lane = threadIdx.x & 31;
    const int half = lane >> 4;
    const int mloc = lane & 15;
    const int tile_m = blockIdx.x * 8 + wave;   // 8192 tiles total

    const int trow = tile_m * 16 + mloc;        // [0, 131072)
    const int b    = trow >> 12;
    const int j    = trow & (NF - 1);
    const float* Arow = tokens + ((size_t)b * NTOK + NC + j) * E;

    v8f acc = {};

    for (int ks = 0; ks < 8; ++ks) {
        const v16h a  = load_a_frag(Arow, ks * 32, half);
        const v16h bf = *(const v16h*)(fragF + ((size_t)ks * 32 + lane) * 16);
        acc = __builtin_amdgcn_wmma_f32_16x16x32_f16(
            false, a, false, bf, (short)0, acc, false, false);
    }

    const float bias = bf_[mloc];
#pragma unroll
    for (int r = 0; r < 8; ++r) {
        const int M = r + half * 8;
        const int t = tile_m * 16 + M;
        const int bb = t >> 12;
        const int jj = t & (NF - 1);
        fine_all[((size_t)bb * NF + jj) * 16 + mloc] = acc[r] + bias;
    }
}

// ---------------------------------------------------------------------------
// Kernel 3: per-batch mask prefix-scan + compaction gather -> fine image.
// One block (256 threads) per batch; each thread owns 16 mask slots.
// patches[b,j] = mask[b,j] ? fine_all[b, rank-1] : 0, pixel-shuffled 4x4.
// ---------------------------------------------------------------------------
__global__ void __launch_bounds__(256)
scan_gather_kernel(const int* __restrict__ mask,
                   const float* __restrict__ fine_all,
                   float* __restrict__ fine_img) {
    __shared__ int sums[256];
    const int b   = blockIdx.x;
    const int tid = threadIdx.x;
    const int base = tid * 16;

    int lm[16];
    int cnt = 0;
#pragma unroll
    for (int c = 0; c < 4; ++c) {
        const v4i mv = *(const v4i*)(mask + (size_t)b * NF + base + c * 4);
#pragma unroll
        for (int i = 0; i < 4; ++i) {
            lm[c * 4 + i] = (mv[i] != 0) ? 1 : 0;
            cnt += lm[c * 4 + i];
        }
    }
    sums[tid] = cnt;
    __syncthreads();
    // Hillis-Steele inclusive scan over the 256 chunk sums
    for (int off = 1; off < 256; off <<= 1) {
        int v = (tid >= off) ? sums[tid - off] : 0;
        __syncthreads();
        sums[tid] += v;
        __syncthreads();
    }
    int run = sums[tid] - cnt;   // exclusive prefix for this chunk

    for (int i = 0; i < 16; ++i) {
        const int j  = base + i;
        const int hf = j >> 6;
        const int wf = j & 63;
        v4f rows[4];
        if (lm[i]) {
            const float* src = fine_all + ((size_t)b * NF + run) * 16;
#pragma unroll
            for (int n = 0; n < 4; ++n) rows[n] = *(const v4f*)(src + n * 4);
            ++run;
        } else {
#pragma unroll
            for (int n = 0; n < 4; ++n) rows[n] = (v4f)0.0f;
        }
#pragma unroll
        for (int dy = 0; dy < 4; ++dy) {
            *(v4f*)(fine_img +
                    ((size_t)b * IMG + (hf * 4 + dy)) * IMG + wf * 4) = rows[dy];
        }
    }
}

// ---------------------------------------------------------------------------
// Kernel 4: fused 3x3 conv (coarse) + sigmoid blend + bilinear 256->32.
// One thread per output element (B*32*32 = 32768).
// ---------------------------------------------------------------------------
__device__ __forceinline__ float sample_blend(const float* __restrict__ cimg,
                                              const float* __restrict__ fimg,
                                              const float* __restrict__ noise,
                                              const float* cw,
                                              int b, int h, int w) {
    float s = 0.0f;
#pragma unroll
    for (int ky = 0; ky < 3; ++ky) {
        const int hh = h + ky - 1;
        if (hh < 0 || hh >= IMG) continue;
#pragma unroll
        for (int kx = 0; kx < 3; ++kx) {
            const int ww = w + kx - 1;
            if (ww < 0 || ww >= IMG) continue;
            s += cw[ky * 3 + kx] *
                 cimg[((size_t)b * IMG + hh) * IMG + ww];
        }
    }
    const size_t off = ((size_t)b * IMG + h) * IMG + w;
    const float fv = fimg[off];
    const float nz = noise[off];
    const float wgt = 1.0f / (1.0f + __expf(-0.1f * nz));
    return s * (1.0f - wgt) + fv * wgt;
}

__global__ void __launch_bounds__(256)
conv_blend_down_kernel(const float* __restrict__ coarse_img,
                       const float* __restrict__ fine_img,
                       const float* __restrict__ noise,
                       const float* __restrict__ conv_w,
                       float* __restrict__ out) {
    const int idx = blockIdx.x * blockDim.x + threadIdx.x;
    if (idx >= BATCH * 32 * 32) return;
    const int b = idx >> 10;
    const int p = (idx >> 5) & 31;
    const int q = idx & 31;

    float cw[9];
#pragma unroll
    for (int i = 0; i < 9; ++i) cw[i] = conv_w[i];

    const double ph = (double)p * 255.0 / 31.0;
    const double pw = (double)q * 255.0 / 31.0;
    const int h0 = (int)floor(ph);
    const int w0 = (int)floor(pw);
    const int h1 = min(h0 + 1, IMG - 1);
    const int w1 = min(w0 + 1, IMG - 1);
    const float fh = (float)(ph - (double)h0);
    const float fw = (float)(pw - (double)w0);

    const float v00 = sample_blend(coarse_img, fine_img, noise, cw, b, h0, w0);
    const float v01 = sample_blend(coarse_img, fine_img, noise, cw, b, h0, w1);
    const float v10 = sample_blend(coarse_img, fine_img, noise, cw, b, h1, w0);
    const float v11 = sample_blend(coarse_img, fine_img, noise, cw, b, h1, w1);

    const float top = v00 * (1.0f - fw) + v01 * fw;
    const float bot = v10 * (1.0f - fw) + v11 * fw;
    out[idx] = top * (1.0f - fh) + bot * fh;
}

// ---------------------------------------------------------------------------
extern "C" void kernel_launch(void* const* d_in, const int* in_sizes, int n_in,
                              void* d_out, int out_size, void* d_ws, size_t ws_size,
                              hipStream_t stream) {
    const float* tokens  = (const float*)d_in[0];   // (32, 5120, 256)
    const int*   mask    = (const int*)  d_in[1];   // (32, 4096)
    const float* Wc      = (const float*)d_in[2];   // (256, 64)
    const float* bc      = (const float*)d_in[3];   // (64,)
    const float* Wf      = (const float*)d_in[4];   // (256, 16)
    const float* bf      = (const float*)d_in[5];   // (16,)
    const float* conv_w  = (const float*)d_in[6];   // (1,1,3,3)
    const float* noise   = (const float*)d_in[7];   // (32,1,256,256)
    float* out = (float*)d_out;                     // (32,1,32,32)

    const size_t imgN = (size_t)BATCH * IMG * IMG;  // 2,097,152 floats
    float* coarse_img = (float*)d_ws;               // 8 MB
    float* fine_img   = coarse_img + imgN;          // 8 MB
    float* fine_all   = fine_img + imgN;            // 8 MB (B*4096*16)
    _Float16* fragC   = (_Float16*)(fine_all + imgN);     // 32 KB
    _Float16* fragF   = fragC + 8 * 4 * 32 * 16;          // 8 KB

    // 0) pack f16 B-fragments (20480 elements -> 80 blocks)
    pack_bfrag_kernel<<<80, 256, 0, stream>>>(Wc, Wf, fragC, fragF);

    // 1) coarse GEMM: 2048 M-tiles, 8 waves/block -> 256 blocks
    coarse_gemm_kernel<<<256, 256, 0, stream>>>(tokens, fragC, bc, coarse_img);

    // 2) fine GEMM: 8192 M-tiles, 8 waves/block -> 1024 blocks
    fine_gemm_kernel<<<1024, 256, 0, stream>>>(tokens, fragF, bf, fine_all);

    // 3) mask scan + gather: one block per batch
    scan_gather_kernel<<<BATCH, 256, 0, stream>>>(mask, fine_all, fine_img);

    // 4) fused conv + blend + bilinear downsample
    conv_blend_down_kernel<<<(BATCH * 32 * 32) / 256, 256, 0, stream>>>(
        coarse_img, fine_img, noise, conv_w, out);
}
